// GraphNet_66460323938747
// MI455X (gfx1250) — compile-verified, matrix operand order checked by reading, not run on hardware
//
#include <hip/hip_runtime.h>

typedef __attribute__((ext_vector_type(16))) _Float16 v16h;
typedef __attribute__((ext_vector_type(8)))  float    v8f;

#define DEV __device__ __forceinline__

DEV int lane_id() { return threadIdx.x & 31; }

// K-index map for 16-bit A (16x32) and B (32x16) fragments on wave32.
DEV int kmap(int e, int half) {
  return (e & 8) ? (16 + (half << 3) + (e & 7)) : ((half << 3) + e);
}

// f32 source fragment load (A-style; also B of X @ W^T). Converts f32->f16.
DEV v16h load_frag_f32(const float* src, int ld, int r0, int c0) {
  int l = lane_id(); int half = l >> 4; int m = l & 15;
  const float* row = src + (long)(r0 + m) * ld + c0;
  v16h out;
#pragma unroll
  for (int e = 0; e < 16; ++e) out[e] = (_Float16)row[kmap(e, half)];
  return out;
}

// f16 source fragment load: per lane, two contiguous 8-half (16B) chunks.
DEV v16h load_frag_h16(const _Float16* src, int ld, int r0, int c0) {
  int l = lane_id(); int half = l >> 4; int m = l & 15;
  const _Float16* row = src + (long)(r0 + m) * ld + c0 + (half << 3);
  v16h out;
#pragma unroll
  for (int e = 0; e < 8; ++e) out[e] = row[e];
#pragma unroll
  for (int e = 0; e < 8; ++e) out[8 + e] = row[16 + e];
  return out;
}

DEV void store_frag_C(float* dst, int ld, int r0, int c0, v8f c) {
  int l = lane_id(); int half = l >> 4; int n = l & 15;
#pragma unroll
  for (int r = 0; r < 8; ++r) dst[(long)(r0 + r + 8 * half) * ld + c0 + n] = c[r];
}

DEV v8f load_frag_C(const float* src, int ld, int r0, int c0) {
  int l = lane_id(); int half = l >> 4; int n = l & 15;
  v8f c;
#pragma unroll
  for (int r = 0; r < 8; ++r) c[r] = src[(long)(r0 + r + 8 * half) * ld + c0 + n];
  return c;
}

// C fragment -> f16 row-major tile (single convert at producer).
DEV void store_frag_h16(_Float16* dst, int ld, int r0, int c0, v8f c) {
  int l = lane_id(); int half = l >> 4; int n = l & 15;
#pragma unroll
  for (int r = 0; r < 8; ++r)
    dst[(long)(r0 + r + 8 * half) * ld + c0 + n] = (_Float16)c[r];
}

// C fragment stored transposed: element (m,n) -> dst[n*ldT + m] (for V^T).
DEV void store_frag_h16_T(_Float16* dst, int ldT, int r0, int c0, v8f c) {
  int l = lane_id(); int half = l >> 4; int n = l & 15;
#pragma unroll
  for (int r = 0; r < 8; ++r)
    dst[(long)(c0 + n) * ldT + r0 + r + 8 * half] = (_Float16)c[r];
}

// Bias folded into accumulator init: C(m,n) = bias[c0+n] (lane-constant).
DEV v8f bias_acc(const float* bias, int c0) {
  float bv = bias[c0 + (lane_id() & 15)];
  v8f c;
#pragma unroll
  for (int r = 0; r < 8; ++r) c[r] = bv;
  return c;
}

DEV v8f wmma16(v16h a, v16h b, v8f c) {
  return __builtin_amdgcn_wmma_f32_16x16x32_f16(false, a, false, b, (short)0, c, false, false);
}

// LayerNorm over one 64-wide row, 4 lanes per row (quad), f32 stats via shuffles.
// x[16] = this lane's quarter (features q*16..q*16+15). Writes f16 row.
DEV void ln64_reg(float* x, const float* g, const float* be, _Float16* dRow,
                  int q, bool reluAfter) {
  float sum = 0.f;
#pragma unroll
  for (int i = 0; i < 16; ++i) sum += x[i];
  sum += __shfl_xor(sum, 1); sum += __shfl_xor(sum, 2);
  float mu = sum * (1.f / 64.f);
  float var = 0.f;
#pragma unroll
  for (int i = 0; i < 16; ++i) { float d = x[i] - mu; var += d * d; }
  var += __shfl_xor(var, 1); var += __shfl_xor(var, 2);
  float inv = rsqrtf(var * (1.f / 64.f) + 1e-12f);
#pragma unroll
  for (int i = 0; i < 16; ++i) {
    int o = q * 16 + i;
    float y = (x[i] - mu) * inv * g[o] + be[o];
    dRow[o] = (_Float16)(reluAfter ? fmaxf(y, 0.f) : y);
  }
}

// ---------------------------------------------------------------------------
// Subgraph kernel: one block (4 waves) per 32-atom subgraph.
// ---------------------------------------------------------------------------
struct SubP {
  const int* atoms_prot; const int* atoms_rna;
  const float* emb;
  const float* w0;  const float* b0;  const float* g0;  const float* be0;
  const _Float16* w0b16; const float* b0b; const float* g0b; const float* be0b;
  const _Float16* Wq16[3]; const _Float16* Wk16[3]; const _Float16* Wv16[3];
  const float* bq[3]; const float* bk[3]; const float* bv[3];
  const float* lg[3]; const float* lb[3];
  float* prot_out; float* rdna_out;
  int nprot;
};

__global__ __launch_bounds__(128)
void subgraph_kernel(SubP P) {
  __shared__ _Float16 sh16[32 * 64];     // h (A operand)
  __shared__ _Float16 sq16[32 * 64];     // q (+bias)
  __shared__ _Float16 sk16[32 * 64];     // k (+bias)
  __shared__ _Float16 svT[64 * 32];      // v^T [n=64][k=32] (+bias)
  __shared__ _Float16 sp16[2 * 32 * 32]; // probs per head
  __shared__ float    ssf[32 * 64];      // f32 scratch: mlp-out / scores / ctx
  int tid = threadIdx.x;
  int w = tid >> 5;
  int row = tid >> 2;       // 0..31 (one quad per row)
  int q = tid & 3;          // quad slot
  int s = blockIdx.x;
  bool isProt = s < P.nprot;
  const int* atoms = isProt ? (P.atoms_prot + (size_t)s * 32)
                            : (P.atoms_rna + (size_t)(s - P.nprot) * 32);
  // --- embed + mlp0 (K=12, scalar, all 128 threads: quad per row) ---
  {
    int a = atoms[row];
    float e[12];
#pragma unroll
    for (int j = 0; j < 12; ++j) e[j] = P.emb[a * 12 + j];
    float x[16];
#pragma unroll
    for (int i = 0; i < 16; ++i) {
      int o = q * 16 + i;
      float acc = P.b0[o];
      const float* wr = P.w0 + o * 12;
#pragma unroll
      for (int j = 0; j < 12; ++j) acc += e[j] * wr[j];
      x[i] = acc;
    }
    ln64_reg(x, P.g0, P.be0, sh16 + row * 64, q, true);
  }
  __syncthreads();
  // --- mlp0b: ssf = h @ W0b^T + b0b (WMMA, bias folded) ---
  for (int mt = 0; mt < 2; ++mt) {
    v8f acc = bias_acc(P.b0b, w * 16);
    for (int k0 = 0; k0 < 64; k0 += 32) {
      v16h A = load_frag_h16(sh16, 64, mt * 16, k0);
      v16h Bf = load_frag_h16(P.w0b16, 64, w * 16, k0);
      acc = wmma16(A, Bf, acc);
    }
    store_frag_C(ssf, 64, mt * 16, w * 16, acc);
  }
  __syncthreads();
  {
    float x[16];
#pragma unroll
    for (int i = 0; i < 16; ++i) x[i] = ssf[row * 64 + q * 16 + i];
    ln64_reg(x, P.g0b, P.be0b, sh16 + row * 64, q, true);
  }
  __syncthreads();
  // --- 3 attention layers ---
  for (int L = 0; L < 3; ++L) {
    // q/k row-major f16, v transposed f16; biases folded into C init.
    for (int mi = 0; mi < 3; ++mi) {
      const _Float16* Wm = (mi == 0) ? P.Wq16[L] : (mi == 1) ? P.Wk16[L] : P.Wv16[L];
      const float* bb = (mi == 0) ? P.bq[L] : (mi == 1) ? P.bk[L] : P.bv[L];
      for (int mt = 0; mt < 2; ++mt) {
        v8f acc = bias_acc(bb, w * 16);
        for (int k0 = 0; k0 < 64; k0 += 32) {
          v16h A = load_frag_h16(sh16, 64, mt * 16, k0);
          v16h Bf = load_frag_h16(Wm, 64, w * 16, k0);
          acc = wmma16(A, Bf, acc);
        }
        if (mi == 2)      store_frag_h16_T(svT, 32, mt * 16, w * 16, acc);
        else if (mi == 1) store_frag_h16(sk16, 64, mt * 16, w * 16, acc);
        else              store_frag_h16(sq16, 64, mt * 16, w * 16, acc);
      }
    }
    __syncthreads();
    // scores (2 heads, hd=32): f32 into ssf
    for (int tt = w; tt < 8; tt += 4) {
      int hh = tt >> 2, mt = (tt >> 1) & 1, nt = tt & 1;
      v16h A = load_frag_h16(sq16, 64, mt * 16, hh * 32);
      v16h Bf = load_frag_h16(sk16, 64, nt * 16, hh * 32);
      v8f acc = {};
      acc = wmma16(A, Bf, acc);
      store_frag_C(ssf + hh * 1024, 32, mt * 16, nt * 16, acc);
    }
    __syncthreads();
    // softmax: 64 rows over 128 threads (pair per row)
    {
      int rr = tid >> 1, hc = tid & 1;
      int hh = rr >> 5, m = rr & 31;
      const float* r = ssf + hh * 1024 + m * 32 + hc * 16;
      _Float16* pr = sp16 + hh * 1024 + m * 32 + hc * 16;
      const float sc = 0.17677669529663687f; // 1/sqrt(32)
      float v[16];
#pragma unroll
      for (int j = 0; j < 16; ++j) v[j] = r[j] * sc;
      float mx = -3.4e38f;
#pragma unroll
      for (int j = 0; j < 16; ++j) mx = fmaxf(mx, v[j]);
      mx = fmaxf(mx, __shfl_xor(mx, 1));
      float sum = 0.f;
#pragma unroll
      for (int j = 0; j < 16; ++j) { v[j] = __expf(v[j] - mx); sum += v[j]; }
      sum += __shfl_xor(sum, 1);
      float inv = 1.f / sum;
#pragma unroll
      for (int j = 0; j < 16; ++j) pr[j] = (_Float16)(v[j] * inv);
    }
    __syncthreads();
    // ctx = probs @ v -> f32 ssf (scores now dead)
    for (int tt = w; tt < 8; tt += 4) {
      int hh = tt >> 2, mt = (tt >> 1) & 1, nt = tt & 1;
      v16h A = load_frag_h16(sp16 + hh * 1024, 32, mt * 16, 0);
      v16h Bf = load_frag_h16(svT, 32, hh * 32 + nt * 16, 0);
      v8f acc = {};
      acc = wmma16(A, Bf, acc);
      store_frag_C(ssf, 64, mt * 16, hh * 32 + nt * 16, acc);
    }
    __syncthreads();
    {
      float x[16];
#pragma unroll
      for (int i = 0; i < 16; ++i) {
        int o = q * 16 + i;
        x[i] = fmaxf(ssf[row * 64 + o], 0.f) + (float)sh16[row * 64 + o];
      }
      ln64_reg(x, P.lg[L], P.lb[L], sh16 + row * 64, q, false);
    }
    __syncthreads();
  }
  // --- max over atoms ---
  if (tid < 64) {
    float mx = -3.4e38f;
#pragma unroll
    for (int l2 = 0; l2 < 32; ++l2) mx = fmaxf(mx, (float)sh16[l2 * 64 + tid]);
    float* out = isProt ? (P.prot_out + (size_t)s * 64)
                        : (P.rdna_out + (size_t)(s - P.nprot) * 64);
    out[tid] = mx;
  }
}

// ---------------------------------------------------------------------------
// Projection: Y[b] = X[b] @ W^T + bias -> f16 (optionally transposed for V).
// ---------------------------------------------------------------------------
__global__ __launch_bounds__(128)
void proj_kernel(const float* X, const _Float16* W16, const float* bias,
                 _Float16* Y, int din, int dout, long xbs, long ybs,
                 int M, int transposeOut) {
  int w = threadIdx.x >> 5;
  long b = blockIdx.y;
  int row0 = blockIdx.x * 32;
  const float* Xb = X + b * xbs + (long)row0 * din;
  _Float16* Yb = Y + b * ybs;
  int ntiles = dout >> 4;
  for (int nt = w; nt < ntiles; nt += 4) {
    for (int mt = 0; mt < 2; ++mt) {
      v8f acc = bias_acc(bias, nt * 16);
      for (int k0 = 0; k0 < din; k0 += 32) {
        v16h A = load_frag_f32(Xb, din, mt * 16, k0);
        v16h Bf = load_frag_h16(W16, din, nt * 16, k0);
        acc = wmma16(A, Bf, acc);
      }
      if (transposeOut)
        store_frag_h16_T(Yb, M, row0 + mt * 16, nt * 16, acc);
      else
        store_frag_h16(Yb + (long)row0 * dout, dout, mt * 16, nt * 16, acc);
    }
  }
}

// ---------------------------------------------------------------------------
// Single-head attention, online softmax over 32-key tiles.
// Q,K: f16 [M][hd]; VT: f16 [hd][Mk] (transposed).  Grid (Mq/32, B).
// ---------------------------------------------------------------------------
__global__ __launch_bounds__(128)
void attn_kernel(const _Float16* Q, const _Float16* K, const _Float16* VT, float* dst,
                 int Mk, int hd, int dstLd, int dstOff, int addRes, float scale,
                 long qbs, long kbs, long dbs) {
  __shared__ _Float16 sQ16[32 * 64];
  __shared__ float    sS[32 * 32];
  __shared__ _Float16 sP16[32 * 32];
  __shared__ float    sA[32 * 64];
  __shared__ float    sM[32];
  __shared__ float    sL[32];
  int tid = threadIdx.x; int w = tid >> 5;
  long b = blockIdx.y;
  int row0 = blockIdx.x * 32;
  int hdLog = (hd == 32) ? 5 : 6;
  const _Float16* Qb = Q + b * qbs + (long)row0 * hd;
  const _Float16* Kb = K + b * kbs;
  const _Float16* VTb = VT + b * kbs;
  int nQ = 32 * hd;
  for (int i = tid; i < nQ; i += 128) { sQ16[i] = Qb[i]; sA[i] = 0.f; }
  if (tid < 32) { sM[tid] = -3.4e38f; sL[tid] = 0.f; }
  __syncthreads();
  int nkt = Mk >> 5;
  for (int kt = 0; kt < nkt; ++kt) {
    const _Float16* Kt = Kb + (long)kt * 32 * hd;
    // prefetch next K tile (contiguous) and next VT tile columns
    if (kt + 1 < nkt) {
      if (tid * 16 < 32 * hd)
        __builtin_prefetch(Kt + 32 * hd + tid * 16, 0, 3);
      if (tid < hd)
        __builtin_prefetch(VTb + (long)tid * Mk + (kt + 1) * 32, 0, 3);
    }
    for (int tt = w; tt < 4; tt += 4) {
      int mt = tt >> 1, nt = tt & 1;
      v8f acc = {};
      for (int k0 = 0; k0 < hd; k0 += 32) {
        v16h A = load_frag_h16(sQ16, hd, mt * 16, k0);
        v16h Bf = load_frag_h16(Kt, hd, nt * 16, k0);
        acc = wmma16(A, Bf, acc);
      }
      store_frag_C(sS, 32, mt * 16, nt * 16, acc);
    }
    __syncthreads();
    // online softmax: quad per row (4 lanes x 8 cols)
    {
      int row = tid >> 2, q = tid & 3;
      const float* r = sS + row * 32 + q * 8;
      _Float16* pr = sP16 + row * 32 + q * 8;
      float v[8];
#pragma unroll
      for (int j = 0; j < 8; ++j) v[j] = r[j] * scale;
      float mx = -3.4e38f;
#pragma unroll
      for (int j = 0; j < 8; ++j) mx = fmaxf(mx, v[j]);
      mx = fmaxf(mx, __shfl_xor(mx, 1));
      mx = fmaxf(mx, __shfl_xor(mx, 2));
      float newm = fmaxf(sM[row], mx);
      float corr = __expf(sM[row] - newm);
      float sum = 0.f;
#pragma unroll
      for (int j = 0; j < 8; ++j) {
        v[j] = __expf(v[j] - newm);
        pr[j] = (_Float16)v[j];
        sum += v[j];
      }
      sum += __shfl_xor(sum, 1);
      sum += __shfl_xor(sum, 2);
      if (q == 0) { sL[row] = sL[row] * corr + sum; sM[row] = newm; }
      // rescale this row's accumulator quarter
      int qw = hd >> 2;
      float* ar = sA + row * hd + q * qw;
      for (int c = 0; c < qw; ++c) ar[c] *= corr;
    }
    __syncthreads();
    int nth = hd >> 4;
    for (int tt = w; tt < 2 * nth; tt += 4) {
      int mt = tt / nth, nt = tt % nth;
      v8f acc = load_frag_C(sA, hd, mt * 16, nt * 16);
      v16h A = load_frag_h16(sP16, 32, mt * 16, 0);
      v16h Bf = load_frag_h16(VTb, Mk, nt * 16, kt * 32);
      acc = wmma16(A, Bf, acc);
      store_frag_C(sA, hd, mt * 16, nt * 16, acc);
    }
    __syncthreads();
  }
  float* Db = dst + b * dbs + (long)row0 * dstLd + dstOff;
  for (int i = tid; i < nQ; i += 128) {
    int m = i >> hdLog, c = i & (hd - 1);
    float v = sA[i] / sL[m];
    float* p = Db + (long)m * dstLd + c;
    *p = addRes ? (*p + v) : v;
  }
}

// f32 -> f16 weight pre-conversion
__global__ void cvtw_kernel(const float* src, _Float16* dst, int n) {
  int i = blockIdx.x * blockDim.x + threadIdx.x;
  if (i < n) dst[i] = (_Float16)src[i];
}

// aa_emb = relu(aa @ aa_W^T) written into inp[:, :, 0:64]
__global__ void aaemb_kernel(const float* aa, const float* Waa, float* inp, int total) {
  int idx = blockIdx.x * blockDim.x + threadIdx.x;
  if (idx >= total) return;
  int r = idx >> 6, o = idx & 63;
  const float* a = aa + (long)r * 20;
  const float* wr = Waa + o * 20;
  float acc = 0.f;
#pragma unroll
  for (int j = 0; j < 20; ++j) acc += a[j] * wr[j];
  inp[(long)r * 128 + o] = fmaxf(acc, 0.f);
}

__global__ void cpy_kernel(const float* prot, float* inp, int total) {
  int idx = blockIdx.x * blockDim.x + threadIdx.x;
  if (idx >= total) return;
  int r = idx >> 6, o = idx & 63;
  inp[(long)r * 128 + 64 + o] = prot[idx];
}

// pooled = max over residues of hs; out[b] = pooled . reg_W
__global__ __launch_bounds__(128)
void pool_kernel(const float* hs, const float* regW, float* out) {
  __shared__ float red[128];
  int b = blockIdx.x, c = threadIdx.x;
  const float* base = hs + (long)b * 1024 * 128 + c;
  float mx = -3.4e38f;
  for (int r = 0; r < 1024; ++r) mx = fmaxf(mx, base[(long)r * 128]);
  red[c] = mx * regW[c];
  __syncthreads();
  for (int s2 = 64; s2 > 0; s2 >>= 1) {
    if (c < s2) red[c] += red[c + s2];
    __syncthreads();
  }
  if (c == 0) out[b] = red[0];
}

// ---------------------------------------------------------------------------
// Input index map: [0]=atom_attributes, [1]=rna_atom_attributes, [2]=aa_attributes,
// then params flattened pytree-style (sorted dict keys):
//  a2l{Wk,Wq,Wv,bk,bq,bv}=3..8, aa_W=9, gg1=10..15, gg2=16..21, l2a=22..27,
//  l2l1=28..33, l2l2=34..39, reg_W=40, sub.emb=41,
//  sub.layers[l]{Wk,Wq,Wv,beta,bk,bq,bv,g}=42+8l, sub.mlp0{W,b,beta,g}=66..69,
//  sub.mlp0b=70..73
// ---------------------------------------------------------------------------
extern "C" void kernel_launch(void* const* d_in, const int* in_sizes, int n_in,
                              void* d_out, int out_size, void* d_ws, size_t ws_size,
                              hipStream_t stream) {
  const int B = 8, N_AA = 1024, N_RNA = 512;
  auto F = [&](int i) { return (const float*)d_in[i]; };
  const int* atom = (const int*)d_in[0];
  const int* rna  = (const int*)d_in[1];
  const float* aa = F(2);

  // f32 region
  float* ws = (float*)d_ws;
  float* prot = ws;
  float* rdna = prot + (size_t)B * N_AA * 64;
  float* inp  = rdna + (size_t)B * N_RNA * 64;
  float* hs   = inp + (size_t)B * N_AA * 128;
  // f16 region
  _Float16* hp = (_Float16*)(hs + (size_t)B * N_AA * 128);
  auto carve = [&](size_t n) { _Float16* p = hp; hp += n; return p; };
  _Float16* Qh  = carve((size_t)B * N_AA * 64);
  _Float16* Kh  = carve((size_t)B * N_AA * 64);
  _Float16* VTh = carve((size_t)B * N_AA * 64);
  _Float16* Q2h = carve((size_t)B * N_RNA * 32);
  _Float16* K2h = carve((size_t)B * N_RNA * 32);
  _Float16* VT2h = carve((size_t)B * N_RNA * 32);
  // f16 weights
  auto cv = [&](const float* s, size_t n) {
    _Float16* d = carve(n);
    cvtw_kernel<<<((int)n + 255) / 256, 256, 0, stream>>>(s, d, (int)n);
    return d;
  };
  _Float16* a2l_q = cv(F(4), 4096);  _Float16* a2l_k = cv(F(3), 4096);  _Float16* a2l_v = cv(F(5), 4096);
  _Float16* l2l1_q = cv(F(29), 2048); _Float16* l2l1_k = cv(F(28), 2048); _Float16* l2l1_v = cv(F(30), 2048);
  _Float16* l2l2_q = cv(F(35), 2048); _Float16* l2l2_k = cv(F(34), 2048); _Float16* l2l2_v = cv(F(36), 2048);
  _Float16* l2a_q = cv(F(23), 4096); _Float16* l2a_k = cv(F(22), 4096); _Float16* l2a_v = cv(F(24), 4096);
  _Float16* gg1_q = cv(F(11), 8192); _Float16* gg1_k = cv(F(10), 8192); _Float16* gg1_v = cv(F(12), 8192);
  _Float16* gg2_q = cv(F(17), 8192); _Float16* gg2_k = cv(F(16), 8192); _Float16* gg2_v = cv(F(18), 8192);
  _Float16* w0b16 = cv(F(70), 4096);
  _Float16* lw16[9];
  for (int l = 0; l < 3; ++l) {
    int base = 42 + 8 * l;
    lw16[l * 3 + 0] = cv(F(base + 1), 4096); // Wq
    lw16[l * 3 + 1] = cv(F(base + 0), 4096); // Wk
    lw16[l * 3 + 2] = cv(F(base + 2), 4096); // Wv
  }

  SubP P;
  P.atoms_prot = atom; P.atoms_rna = rna;
  P.emb = F(41);
  P.w0 = F(66); P.b0 = F(67); P.be0 = F(68); P.g0 = F(69);
  P.w0b16 = w0b16; P.b0b = F(71); P.be0b = F(72); P.g0b = F(73);
  for (int l = 0; l < 3; ++l) {
    int base = 42 + 8 * l;
    P.Wq16[l] = lw16[l * 3 + 0]; P.Wk16[l] = lw16[l * 3 + 1]; P.Wv16[l] = lw16[l * 3 + 2];
    P.lb[l] = F(base + 3);
    P.bk[l] = F(base + 4); P.bq[l] = F(base + 5); P.bv[l] = F(base + 6);
    P.lg[l] = F(base + 7);
  }
  P.prot_out = prot; P.rdna_out = rdna; P.nprot = B * N_AA;
  subgraph_kernel<<<dim3(B * (N_AA + N_RNA)), dim3(128), 0, stream>>>(P);

  aaemb_kernel<<<(B * N_AA * 64 + 255) / 256, 256, 0, stream>>>(aa, F(9), inp, B * N_AA * 64);

  long s64p = (long)N_AA * 64, s64r = (long)N_RNA * 64;
  long s32r = (long)N_RNA * 32, s128 = (long)N_AA * 128;
  const float kInvS32 = 0.17677669529663687f;

  // a2l: rdna += attn(rdna, prot), hd=64
  proj_kernel<<<dim3(N_RNA / 32, B), 128, 0, stream>>>(rdna, a2l_q, F(7), Qh, 64, 64, s64r, s64r, N_RNA, 0);
  proj_kernel<<<dim3(N_AA / 32, B), 128, 0, stream>>>(prot, a2l_k, F(6), Kh, 64, 64, s64p, s64p, N_AA, 0);
  proj_kernel<<<dim3(N_AA / 32, B), 128, 0, stream>>>(prot, a2l_v, F(8), VTh, 64, 64, s64p, s64p, N_AA, 1);
  attn_kernel<<<dim3(N_RNA / 32, B), 128, 0, stream>>>(Qh, Kh, VTh, rdna, N_AA, 64, 64, 0, 1,
                                                       0.125f, s64r, s64p, s64r);
  // l2l: rdna += concat(attn(rdna,rdna,l2l1), attn(rdna,rdna,l2l2)), hd=32 each.
  proj_kernel<<<dim3(N_RNA / 32, B), 128, 0, stream>>>(rdna, l2l1_q, F(32), Qh, 64, 32, s64r, s32r, N_RNA, 0);
  proj_kernel<<<dim3(N_RNA / 32, B), 128, 0, stream>>>(rdna, l2l1_k, F(31), Kh, 64, 32, s64r, s32r, N_RNA, 0);
  proj_kernel<<<dim3(N_RNA / 32, B), 128, 0, stream>>>(rdna, l2l1_v, F(33), VTh, 64, 32, s64r, s32r, N_RNA, 1);
  proj_kernel<<<dim3(N_RNA / 32, B), 128, 0, stream>>>(rdna, l2l2_q, F(38), Q2h, 64, 32, s64r, s32r, N_RNA, 0);
  proj_kernel<<<dim3(N_RNA / 32, B), 128, 0, stream>>>(rdna, l2l2_k, F(37), K2h, 64, 32, s64r, s32r, N_RNA, 0);
  proj_kernel<<<dim3(N_RNA / 32, B), 128, 0, stream>>>(rdna, l2l2_v, F(39), VT2h, 64, 32, s64r, s32r, N_RNA, 1);
  attn_kernel<<<dim3(N_RNA / 32, B), 128, 0, stream>>>(Qh, Kh, VTh, rdna, N_RNA, 32, 64, 0, 1,
                                                       kInvS32, s32r, s32r, s64r);
  attn_kernel<<<dim3(N_RNA / 32, B), 128, 0, stream>>>(Q2h, K2h, VT2h, rdna, N_RNA, 32, 64, 32, 1,
                                                       kInvS32, s32r, s32r, s64r);
  // l2a: prot += attn(prot, rdna), hd=64
  proj_kernel<<<dim3(N_AA / 32, B), 128, 0, stream>>>(prot, l2a_q, F(26), Qh, 64, 64, s64p, s64p, N_AA, 0);
  proj_kernel<<<dim3(N_RNA / 32, B), 128, 0, stream>>>(rdna, l2a_k, F(25), Kh, 64, 64, s64r, s64r, N_RNA, 0);
  proj_kernel<<<dim3(N_RNA / 32, B), 128, 0, stream>>>(rdna, l2a_v, F(27), VTh, 64, 64, s64r, s64r, N_RNA, 1);
  attn_kernel<<<dim3(N_AA / 32, B), 128, 0, stream>>>(Qh, Kh, VTh, prot, N_RNA, 64, 64, 0, 1,
                                                      0.125f, s64p, s64r, s64p);
  // inp[:, :, 64:128] = prot
  cpy_kernel<<<(B * N_AA * 64 + 255) / 256, 256, 0, stream>>>(prot, inp, B * N_AA * 64);

  // gg1 -> hs[:, :, 0:64]; gg2 -> hs[:, :, 64:128]  (din=128, hd=64, no residual)
  proj_kernel<<<dim3(N_AA / 32, B), 128, 0, stream>>>(inp, gg1_q, F(14), Qh, 128, 64, s128, s64p, N_AA, 0);
  proj_kernel<<<dim3(N_AA / 32, B), 128, 0, stream>>>(inp, gg1_k, F(13), Kh, 128, 64, s128, s64p, N_AA, 0);
  proj_kernel<<<dim3(N_AA / 32, B), 128, 0, stream>>>(inp, gg1_v, F(15), VTh, 128, 64, s128, s64p, N_AA, 1);
  attn_kernel<<<dim3(N_AA / 32, B), 128, 0, stream>>>(Qh, Kh, VTh, hs, N_AA, 64, 128, 0, 0,
                                                      0.125f, s64p, s64p, s128);
  proj_kernel<<<dim3(N_AA / 32, B), 128, 0, stream>>>(inp, gg2_q, F(20), Qh, 128, 64, s128, s64p, N_AA, 0);
  proj_kernel<<<dim3(N_AA / 32, B), 128, 0, stream>>>(inp, gg2_k, F(19), Kh, 128, 64, s128, s64p, N_AA, 0);
  proj_kernel<<<dim3(N_AA / 32, B), 128, 0, stream>>>(inp, gg2_v, F(21), VTh, 128, 64, s128, s64p, N_AA, 1);
  attn_kernel<<<dim3(N_AA / 32, B), 128, 0, stream>>>(Qh, Kh, VTh, hs, N_AA, 64, 128, 64, 0,
                                                      0.125f, s64p, s64p, s128);

  pool_kernel<<<B, 128, 0, stream>>>(hs, F(40), (float*)d_out);
}
